// Application_85469849191065
// MI455X (gfx1250) — matrix operand order repaired and checked by reading, not compiled, not run on hardware
//
#include <hip/hip_runtime.h>
#include <hip/hip_bf16.h>

#define NS   4096
#define LAT  64
#define CONDD 219
#define PH2  40
#define EXPE 8
#define HID  512
#define OUTD 171
#define GINN 104
#define K0P  288   // 283 padded to 32
#define XS   40    // LDS row stride (bf16 elems): 80B, 16B aligned, conflict-free

typedef __attribute__((ext_vector_type(16))) __bf16 v16bf;
typedef __attribute__((ext_vector_type(8)))  float  v8f;

static __device__ __forceinline__ unsigned short f2bf(float f) {
  unsigned int u = __float_as_uint(f);
  unsigned int r = u + 0x7FFFu + ((u >> 16) & 1u);  // RNE
  return (unsigned short)(r >> 16);
}
static __device__ __forceinline__ float eluf(float x) {
  return x > 0.f ? x : (__expf(x) - 1.f);
}

// ---------------- gate network: [N,104] -> 128 -> 128 -> softmax(8) ----------
__global__ __launch_bounds__(256) void gate_kernel(
    const float* __restrict__ latent, const float* __restrict__ phase,
    const float* __restrict__ gw1, const float* __restrict__ gb1,
    const float* __restrict__ gw2, const float* __restrict__ gb2,
    const float* __restrict__ gw3, const float* __restrict__ gb3,
    float* __restrict__ coeff) {
  __shared__ float in_s[16][GINN];
  __shared__ float h1[16][128];
  __shared__ float h2[16][128];
  __shared__ float lg[16][8];
  int t = threadIdx.x;
  int n0 = blockIdx.x * 16;
  for (int i = t; i < 16 * GINN; i += 256) {
    int s = i / GINN, c = i % GINN;
    int n = n0 + s;
    in_s[s][c] = (c < PH2) ? phase[n * PH2 + c] : latent[n * LAT + (c - PH2)];
  }
  __syncthreads();
  for (int p = t; p < 16 * 128; p += 256) {
    int s = p >> 7, o = p & 127;
    float acc = gb1[o];
    for (int i = 0; i < GINN; ++i) acc += in_s[s][i] * gw1[i * 128 + o];
    h1[s][o] = eluf(acc);
  }
  __syncthreads();
  for (int p = t; p < 16 * 128; p += 256) {
    int s = p >> 7, o = p & 127;
    float acc = gb2[o];
    for (int i = 0; i < 128; ++i) acc += h1[s][i] * gw2[i * 128 + o];
    h2[s][o] = eluf(acc);
  }
  __syncthreads();
  if (t < 16 * 8) {
    int s = t >> 3, e = t & 7;
    float acc = gb3[e];
    for (int i = 0; i < 128; ++i) acc += h2[s][i] * gw3[i * 8 + e];
    lg[s][e] = acc;
  }
  __syncthreads();
  if (t < 16) {
    float m = lg[t][0];
    for (int e = 1; e < 8; ++e) m = fmaxf(m, lg[t][e]);
    float ex[8], sum = 0.f;
    for (int e = 0; e < 8; ++e) { ex[e] = __expf(lg[t][e] - m); sum += ex[e]; }
    float inv = 1.f / sum;
    int n = n0 + t;
    for (int e = 0; e < 8; ++e) coeff[n * 8 + e] = ex[e] * inv;
  }
}

// ------- build x0 = bf16 concat(cond, latent, pad) and latent cols of x1 -----
__global__ void build_x_kernel(const float* __restrict__ cond,
                               const float* __restrict__ lat,
                               unsigned short* __restrict__ x0,
                               unsigned short* __restrict__ x1) {
  int idx = blockIdx.x * blockDim.x + threadIdx.x;
  int total = NS * (K0P + LAT);
  if (idx >= total) return;
  int n = idx / (K0P + LAT);
  int c = idx % (K0P + LAT);
  if (c < K0P) {
    float v = 0.f;
    if (c < CONDD) v = cond[n * CONDD + c];
    else if (c < CONDD + LAT) v = lat[n * LAT + (c - CONDD)];
    x0[n * K0P + c] = f2bf(v);
  } else {
    int cl = c - K0P;
    x1[n * 576 + 512 + cl] = f2bf(lat[n * LAT + cl]);
  }
}

// ------- transpose+pad weights: W[E][K][O] fp32 -> Wt[E][Opad][Kpad] bf16 ----
__global__ void transpose_w_kernel(const float* __restrict__ W,
                                   unsigned short* __restrict__ Wt,
                                   int K, int O, int Kpad, int Opad) {
  long long idx = (long long)blockIdx.x * blockDim.x + threadIdx.x;
  long long total = (long long)EXPE * Opad * Kpad;
  if (idx >= total) return;
  int k = (int)(idx % Kpad);
  long long t2 = idx / Kpad;
  int o = (int)(t2 % Opad);
  int e = (int)(t2 / Opad);
  float v = 0.f;
  if (k < K && o < O) v = W[((long long)e * K + k) * O + o];
  Wt[idx] = f2bf(v);
}

// --------------------- fused expert-blended GEMM layer -----------------------
// out[n,o] = act( sum_e coeff[n,e] * ( sum_k X[n,k] Wt[e,o,k] + b[e,o] ) )
// MODE 0: ELU -> bf16 to Ybf ; MODE 1: fp32 to Yf32 (cols < OUTD)
template <int MODE>
__global__ __launch_bounds__(256) void moe_layer_kernel(
    const unsigned short* __restrict__ X, int Kpad,
    const unsigned short* __restrict__ Wt, int Opad,
    const float* __restrict__ bias, int Obias,
    const float* __restrict__ coeff,
    unsigned short* __restrict__ Ybf, int Ystride,
    float* __restrict__ Yf32) {
  __shared__ __align__(16) unsigned short xlds[64 * XS];
  __shared__ __align__(16) unsigned short wlds[64 * XS];
  __shared__ float clds[64 * 8];
  __shared__ float blds[8 * 64];

  int t = threadIdx.x;
  int rowBase = blockIdx.x * 64;
  int colBase = blockIdx.y * 64;
  for (int i = t; i < 512; i += 256) clds[i] = coeff[rowBase * 8 + i];
  for (int i = t; i < 512; i += 256) {
    int e = i >> 6, c = i & 63;
    int g = colBase + c;
    blds[i] = (g < Obias) ? bias[e * Obias + g] : 0.f;
  }

  int wave = t >> 5, lane = t & 31;
  int mt = wave & 3;
  int nt0 = (wave >> 2) * 2, nt1 = nt0 + 1;
  int mm = lane & 15, hh = lane >> 4;
  int lr = t >> 2, lq = t & 3;  // cooperative tile load: row, 16B quarter

  v8f f0 = {}, f1 = {};
  int nKT = Kpad >> 5;
  for (int e = 0; e < EXPE; ++e) {
    v8f a0 = {}, a1 = {};
    const unsigned short* We = Wt + (long long)e * Opad * Kpad;
    for (int kt = 0; kt < nKT; ++kt) {
      __syncthreads();
      int gk = (kt << 5) + lq * 8;
      *(uint4*)&xlds[lr * XS + lq * 8] =
          *(const uint4*)&X[(long long)(rowBase + lr) * Kpad + gk];
      *(uint4*)&wlds[lr * XS + lq * 8] =
          *(const uint4*)&We[(long long)(colBase + lr) * Kpad + gk];
      if (kt + 1 < nKT)
        __builtin_prefetch(&We[(long long)(colBase + lr) * Kpad + gk + 32], 0, 3);
      __syncthreads();
      union { unsigned int u[8]; v16bf v; } fa, fb0, fb1;
#pragma unroll
      for (int vv = 0; vv < 8; ++vv) {
        int ka = 2 * vv + 16 * hh;  // A layout
        fa.u[vv]  = *(const unsigned int*)&xlds[(mt * 16 + mm) * XS + ka];
        int kb = 2 * vv + 16 * hh;                                   // B layout
        fb0.u[vv] = *(const unsigned int*)&wlds[(nt0 * 16 + mm) * XS + kb];
        fb1.u[vv] = *(const unsigned int*)&wlds[(nt1 * 16 + mm) * XS + kb];
      }
      a0 = __builtin_amdgcn_wmma_f32_16x16x32_bf16(false, fa.v, false, fb0.v,
                                                   (short)0, a0, false, false);
      a1 = __builtin_amdgcn_wmma_f32_16x16x32_bf16(false, fa.v, false, fb1.v,
                                                   (short)0, a1, false, false);
    }
    float b0v = blds[e * 64 + nt0 * 16 + mm];
    float b1v = blds[e * 64 + nt1 * 16 + mm];
#pragma unroll
    for (int r = 0; r < 8; ++r) {
      float sc = clds[(mt * 16 + r + 8 * hh) * 8 + e];
      f0[r] += sc * (a0[r] + b0v);
      f1[r] += sc * (a1[r] + b1v);
    }
  }

  int col0 = colBase + nt0 * 16 + mm;
  int col1 = colBase + nt1 * 16 + mm;
#pragma unroll
  for (int r = 0; r < 8; ++r) {
    int row = rowBase + mt * 16 + r + 8 * hh;
    if (MODE == 0) {
      Ybf[(long long)row * Ystride + col0] = f2bf(eluf(f0[r]));
      Ybf[(long long)row * Ystride + col1] = f2bf(eluf(f1[r]));
    } else {
      if (col0 < OUTD) Yf32[(long long)row * OUTD + col0] = f0[r];
      if (col1 < OUTD) Yf32[(long long)row * OUTD + col1] = f1[r];
    }
  }
}

extern "C" void kernel_launch(void* const* d_in, const int* in_sizes, int n_in,
                              void* d_out, int out_size, void* d_ws, size_t ws_size,
                              hipStream_t stream) {
  const float* latent    = (const float*)d_in[0];
  const float* condition = (const float*)d_in[1];
  const float* phase     = (const float*)d_in[2];
  const float* gw1 = (const float*)d_in[3];
  const float* gb1 = (const float*)d_in[4];
  const float* gw2 = (const float*)d_in[5];
  const float* gb2 = (const float*)d_in[6];
  const float* gw3 = (const float*)d_in[7];
  const float* gb3 = (const float*)d_in[8];
  const float* w0 = (const float*)d_in[9];
  const float* b0 = (const float*)d_in[10];
  const float* w1 = (const float*)d_in[11];
  const float* b1 = (const float*)d_in[12];
  const float* w2 = (const float*)d_in[13];
  const float* b2 = (const float*)d_in[14];

  float* out   = (float*)d_out;
  float* coeff = out + (size_t)NS * OUTD;  // second output, also blend input

  char* ws = (char*)d_ws;
  size_t off = 0;
  auto alloc = [&](size_t bytes) {
    size_t p = off; off += (bytes + 255) & ~(size_t)255; return p;
  };
  unsigned short* x0  = (unsigned short*)(ws + alloc((size_t)NS * K0P * 2));
  unsigned short* x1  = (unsigned short*)(ws + alloc((size_t)NS * 576 * 2));
  unsigned short* x2  = (unsigned short*)(ws + alloc((size_t)NS * 512 * 2));
  unsigned short* Wt0 = (unsigned short*)(ws + alloc((size_t)EXPE * 512 * K0P * 2));
  unsigned short* Wt1 = (unsigned short*)(ws + alloc((size_t)EXPE * 512 * 576 * 2));
  unsigned short* Wt2 = (unsigned short*)(ws + alloc((size_t)EXPE * 192 * 512 * 2));

  gate_kernel<<<NS / 16, 256, 0, stream>>>(latent, phase, gw1, gb1, gw2, gb2,
                                           gw3, gb3, coeff);
  {
    int total = NS * (K0P + LAT);
    build_x_kernel<<<(total + 255) / 256, 256, 0, stream>>>(condition, latent, x0, x1);
  }
  {
    long long tot = (long long)EXPE * 512 * K0P;
    transpose_w_kernel<<<(int)((tot + 255) / 256), 256, 0, stream>>>(
        w0, Wt0, CONDD + LAT, 512, K0P, 512);
    tot = (long long)EXPE * 512 * 576;
    transpose_w_kernel<<<(int)((tot + 255) / 256), 256, 0, stream>>>(
        w1, Wt1, HID + LAT, 512, 576, 512);
    tot = (long long)EXPE * 192 * 512;
    transpose_w_kernel<<<(int)((tot + 255) / 256), 256, 0, stream>>>(
        w2, Wt2, HID, OUTD, 512, 192);
  }
  moe_layer_kernel<0><<<dim3(NS / 64, 8), 256, 0, stream>>>(
      x0, K0P, Wt0, 512, b0, 512, coeff, x1, 576, nullptr);
  moe_layer_kernel<0><<<dim3(NS / 64, 8), 256, 0, stream>>>(
      x1, 576, Wt1, 512, b1, 512, coeff, x2, 512, nullptr);
  moe_layer_kernel<1><<<dim3(NS / 64, 3), 256, 0, stream>>>(
      x2, 512, Wt2, 192, b2, OUTD, coeff, nullptr, 0, out);
}
